// Informer_33019708572250
// MI455X (gfx1250) — matrix operand order repaired and checked by reading, not compiled, not run on hardware
//
#include <hip/hip_runtime.h>
#include <cmath>
#include <cstdint>
#include <algorithm>

// ---------------- model constants ----------------
#define BSZ     16
#define L_ENC   2048
#define L_DEC   1024
#define PREDL   512
#define C_IN    7
#define C_OUT   7
#define MARKN   4
#define DMODEL  512
#define NHEADS  8
#define DH      64
#define DFF     2048
#define UPAD    48          // max u (40) padded to multiple of 16
#define BHN     (BSZ*NHEADS)
#define NEGINF  (-1.0e9f)

typedef __attribute__((ext_vector_type(16))) _Float16 v16h;
typedef __attribute__((ext_vector_type(8)))  float    v8f;

__device__ __forceinline__ float gelu_exact(float x) {
  return 0.5f * x * (1.0f + erff(x * 0.70710678118654752f));
}

__device__ __forceinline__ unsigned pcg_hash(unsigned x) {
  x = x * 747796405u + 2891336453u;
  unsigned w = ((x >> ((x >> 28) + 4u)) ^ x) * 277803737u;
  return (w >> 22) ^ w;
}
__device__ __forceinline__ unsigned rnd3(unsigned s, unsigned a, unsigned b) {
  return pcg_hash(s * 0x9E3779B9u ^ pcg_hash(a * 0x85EBCA6Bu ^ pcg_hash(b ^ 0x165667B1u)));
}

// -------- CDNA5 async global->LDS copy (ASYNCcnt-tracked) ------------
typedef __attribute__((address_space(3))) unsigned char lds_byte_t;
__device__ __forceinline__ unsigned lds_u32(void* p) {
  // addrspacecast generic->LDS, then ptrtoint gives the 32-bit LDS offset
  return (unsigned)(unsigned long long)(lds_byte_t*)p;
}
__device__ __forceinline__ void async_copy_b128(unsigned lds_off, unsigned gvoff_bytes,
                                                const float* sbase) {
  asm volatile("global_load_async_to_lds_b128 %0, %1, %2"
               :
               : "v"(lds_off), "v"(gvoff_bytes), "s"(sbase)
               : "memory");
}
__device__ __forceinline__ void wait_async6() {
  asm volatile("s_wait_asynccnt 0x6" ::: "memory");
}
__device__ __forceinline__ void wait_async0() {
  asm volatile("s_wait_asynccnt 0x0" ::: "memory");
}

// =====================================================================
// Tiled WMMA GEMM with async-to-LDS double buffering.
// Requires M%64==0, N%128==0, K%32==0.  C = ACT(A@B + bias) [+ resid]
// Block: 256 thr (8 waves, 2(M) x 4(N)); block tile 64x128, K-step 32.
// Each wave: 32x32 output patch = 4 x v_wmma_f32_16x16x32_f16.
// =====================================================================
#define BM 64
#define BN 128
#define BK 32

template<int ACT, bool HAS_RES>
__global__ __launch_bounds__(256)
void wmma_gemm_tiled_k(const float* __restrict__ A, const float* __restrict__ Bm,
                       const float* __restrict__ bias, const float* __restrict__ Rs,
                       float* __restrict__ C, int M, int N, int K)
{
  __shared__ float As[2][BM * BK];   // 64x32 f32
  __shared__ float Bs[2][BK * BN];   // 32x128 f32

  const int tid  = threadIdx.x;
  const int lane = tid & 31, wave = tid >> 5;
  const int wm = wave & 1, wn = wave >> 1;        // 2x4 wave grid
  const int l15 = lane & 15, half = lane >> 4;

  const int tilesN = N / BN;
  const int tm = (blockIdx.x / tilesN) * BM;
  const int tn = (blockIdx.x % tilesN) * BN;

  // issue one K-tile of async copies (6 x b128 per thread)
  auto issue = [&](int buf, int k0) {
#pragma unroll
    for (int t = 0; t < 2; ++t) {                 // A: 512 chunks of 16B
      int c   = tid + t * 256;
      int row = c >> 3, col = (c & 7) << 2;
      unsigned lo = lds_u32(&As[buf][row * BK + col]);
      unsigned go = (unsigned)(((long)(tm + row) * K + k0 + col) * 4);
      async_copy_b128(lo, go, A);
    }
#pragma unroll
    for (int t = 0; t < 4; ++t) {                 // B: 1024 chunks of 16B
      int c   = tid + t * 256;
      int row = c >> 5, col = (c & 31) << 2;
      unsigned lo = lds_u32(&Bs[buf][row * BN + col]);
      unsigned go = (unsigned)(((long)(k0 + row) * N + tn + col) * 4);
      async_copy_b128(lo, go, Bm);
    }
  };

  v8f acc00{}, acc01{}, acc10{}, acc11{};
  const int nk = K / BK;
  issue(0, 0);
  for (int kt = 0; kt < nk; ++kt) {
    int buf = kt & 1;
    if (kt + 1 < nk) { issue(buf ^ 1, (kt + 1) * BK); wait_async6(); }
    else             { wait_async0(); }
    __syncthreads();                               // all waves' copies visible

    const float* Ab_ = As[buf];
    const float* Bb_ = Bs[buf];
    v16h a0, a1, b0, b1;
#pragma unroll
    for (int i = 0; i < 8; ++i) {
      // A 16x32 f16 fragment layout: K = half*16 + 2i
      int kb = (half << 4) + (i << 1);
      int r0 = (wm * 32 + l15) * BK + kb;
      int r1 = (wm * 32 + 16 + l15) * BK + kb;
      a0[2*i]   = (_Float16)Ab_[r0];
      a0[2*i+1] = (_Float16)Ab_[r0 + 1];
      a1[2*i]   = (_Float16)Ab_[r1];
      a1[2*i+1] = (_Float16)Ab_[r1 + 1];
    }
#pragma unroll
    for (int j = 0; j < 8; ++j) {
      // B 32x16 f16 fragment layout: K = half*16 + 2j
      int kk = (half << 4) + (j << 1);
      int c0 = wn * 32 + l15;
      b0[2*j]   = (_Float16)Bb_[kk * BN + c0];
      b0[2*j+1] = (_Float16)Bb_[(kk + 1) * BN + c0];
      b1[2*j]   = (_Float16)Bb_[kk * BN + c0 + 16];
      b1[2*j+1] = (_Float16)Bb_[(kk + 1) * BN + c0 + 16];
    }
    acc00 = __builtin_amdgcn_wmma_f32_16x16x32_f16(false, a0, false, b0, (short)0, acc00, false, false);
    acc01 = __builtin_amdgcn_wmma_f32_16x16x32_f16(false, a0, false, b1, (short)0, acc01, false, false);
    acc10 = __builtin_amdgcn_wmma_f32_16x16x32_f16(false, a1, false, b0, (short)0, acc10, false, false);
    acc11 = __builtin_amdgcn_wmma_f32_16x16x32_f16(false, a1, false, b1, (short)0, acc11, false, false);
    __syncthreads();                               // safe to overwrite this buffer next iter
  }

  // epilogue
  auto store_tile = [&](v8f& acc, int rbase, int col) {
    float bs = bias[col];
#pragma unroll
    for (int r = 0; r < 8; ++r) {
      int mm = rbase + r + (half << 3);
      float v = acc[r] + bs;
      if (ACT == 1) v = gelu_exact(v);
      if (HAS_RES) v += Rs[(long)mm * N + col];
      C[(long)mm * N + col] = v;
    }
  };
  int r0 = tm + wm * 32;
  int c0 = tn + wn * 32 + l15;
  store_tile(acc00, r0,      c0);
  store_tile(acc01, r0,      c0 + 16);
  store_tile(acc10, r0 + 16, c0);
  store_tile(acc11, r0 + 16, c0 + 16);
}

// =====================================================================
// Fallback per-wave WMMA GEMM (irregular shapes / batched / transposed)
// C = ACT( alpha * A @ op(B) + bias ) [+ resid]
// =====================================================================
template<int ACT, bool HAS_BIAS, bool HAS_RES, bool TRANSB>
__global__ __launch_bounds__(256)
void wmma_gemm_k(const float* __restrict__ A, const float* __restrict__ Bm,
                 const float* __restrict__ bias, const float* __restrict__ Rs,
                 float* __restrict__ C,
                 int M, int N, int K, int lda, int ldb, int ldc,
                 long sAo, long sAi, long sBo, long sBi, long sCo, long sCi,
                 int innerCnt, int tilesTotal, float alpha)
{
  int wave = threadIdx.x >> 5;
  int lane = threadIdx.x & 31;
  int tile = blockIdx.x * (blockDim.x >> 5) + wave;
  if (tile >= tilesTotal) return;                 // wave-uniform exit
  int bb    = blockIdx.y;
  int outer = bb / innerCnt, inner = bb % innerCnt;
  const float* Ab = A  + (long)outer * sAo + (long)inner * sAi;
  const float* Bb = Bm + (long)outer * sBo + (long)inner * sBi;
  float*       Cb = C  + (long)outer * sCo + (long)inner * sCi;
  const float* Rb = HAS_RES ? (Rs + (long)outer * sCo + (long)inner * sCi) : nullptr;

  int tilesN = (N + 15) >> 4;
  int tm = (tile / tilesN) << 4;
  int tn = (tile % tilesN) << 4;
  int l15 = lane & 15, half = lane >> 4;
  int am = tm + l15;
  int bn = tn + l15;
  bool mok = am < M, nok = bn < N;
  const float* arow = Ab + (long)am * lda;

  v8f acc{};
  for (int k0 = 0; k0 < K; k0 += 32) {
    v16h av, bv;
#pragma unroll
    for (int i = 0; i < 8; ++i) {
      int kb = k0 + (half << 4) + (i << 1);
      float x0 = mok ? arow[kb]     : 0.0f;
      float x1 = mok ? arow[kb + 1] : 0.0f;
      av[2*i]   = (_Float16)x0;
      av[2*i+1] = (_Float16)x1;
    }
#pragma unroll
    for (int j = 0; j < 8; ++j) {
      int kk = k0 + (half << 4) + (j << 1);
      float y0, y1;
      if (TRANSB) {
        const float* bp = Bb + (long)bn * ldb + kk;
        y0 = nok ? bp[0] : 0.0f;
        y1 = nok ? bp[1] : 0.0f;
      } else {
        y0 = nok ? Bb[(long)kk       * ldb + bn] : 0.0f;
        y1 = nok ? Bb[(long)(kk + 1) * ldb + bn] : 0.0f;
      }
      bv[2*j]   = (_Float16)y0;
      bv[2*j+1] = (_Float16)y1;
    }
    acc = __builtin_amdgcn_wmma_f32_16x16x32_f16(false, av, false, bv, (short)0, acc, false, false);
  }
  if (nok) {
    float bs = HAS_BIAS ? bias[bn] : 0.0f;
#pragma unroll
    for (int r = 0; r < 8; ++r) {
      int mm = tm + r + (half << 3);
      if (mm < M) {
        float v = alpha * acc[r] + bs;
        if (ACT == 1) v = gelu_exact(v);
        if (HAS_RES) v += Rb[(long)mm * ldc + bn];
        Cb[(long)mm * ldc + bn] = v;
      }
    }
  }
}

// =====================================================================
// Circular conv (kernel=3, 512->512) as implicit-im2col WMMA GEMM, K=1536
// =====================================================================
__global__ __launch_bounds__(256)
void wmma_convgemm_k(const float* __restrict__ X, const float* __restrict__ W,
                     const float* __restrict__ bias, float* __restrict__ C,
                     int L, int tilesTotal)
{
  int wave = threadIdx.x >> 5, lane = threadIdx.x & 31;
  int tile = blockIdx.x * 8 + wave;
  if (tile >= tilesTotal) return;
  const int N = DMODEL, K = 3 * DMODEL;
  int tilesN = N >> 4;
  int tm = (tile / tilesN) << 4, tn = (tile % tilesN) << 4;
  int l15 = lane & 15, half = lane >> 4;
  int m = tm + l15;
  int b = m / L, l = m % L;
  int bn = tn + l15;
  v8f acc{};
  for (int k0 = 0; k0 < K; k0 += 32) {            // 512%32==0 -> chunk stays in one tap
    int t   = k0 >> 9;
    int ci0 = k0 & 511;
    int lw  = l + t - 1;
    lw = (lw < 0) ? (L - 1) : (lw >= L ? 0 : lw); // circular pad
    const float* arow = X + ((long)(b * L + lw) * DMODEL + ci0);
    v16h av, bv;
#pragma unroll
    for (int i = 0; i < 8; ++i) {
      int kb = (half << 4) + (i << 1);
      av[2*i]   = (_Float16)arow[kb];
      av[2*i+1] = (_Float16)arow[kb + 1];
    }
#pragma unroll
    for (int j = 0; j < 8; ++j) {
      int kk = k0 + (half << 4) + (j << 1);
      bv[2*j]   = (_Float16)W[(long)kk       * N + bn];
      bv[2*j+1] = (_Float16)W[(long)(kk + 1) * N + bn];
    }
    acc = __builtin_amdgcn_wmma_f32_16x16x32_f16(false, av, false, bv, (short)0, acc, false, false);
  }
  float bs = bias[bn];
#pragma unroll
  for (int r = 0; r < 8; ++r) {
    int mm = tm + r + (half << 3);
    C[(long)mm * N + bn] = acc[r] + bs;
  }
}

// =====================================================================
// Embedding: circular conv (Cin->512) + sinusoidal pos + mark @ time_W + b
// =====================================================================
__global__ void embed_k(const float* __restrict__ x, const float* __restrict__ mark,
                        const float* __restrict__ convW, const float* __restrict__ timeW,
                        const float* __restrict__ timeb, float* __restrict__ out, int L)
{
  long idx = (long)blockIdx.x * blockDim.x + threadIdx.x;
  long total = (long)BSZ * L * DMODEL;
  if (idx >= total) return;
  int  dm = (int)(idx % DMODEL);
  long bl = idx / DMODEL;
  int  l  = (int)(bl % L);
  int  b  = (int)(bl / L);
  float acc = timeb[dm];
#pragma unroll
  for (int t = 0; t < 3; ++t) {
    int lw = l + t - 1;
    lw = (lw < 0) ? (L - 1) : (lw >= L ? 0 : lw);
    const float* xr = x + (long)(b * L + lw) * C_IN;
#pragma unroll
    for (int ci = 0; ci < C_IN; ++ci)
      acc += xr[ci] * convW[(t * C_IN + ci) * DMODEL + dm];
  }
  const float* mr = mark + bl * MARKN;
#pragma unroll
  for (int j = 0; j < MARKN; ++j) acc += mr[j] * timeW[j * DMODEL + dm];
  int i2 = dm >> 1;
  float div = expf(-(float)(2 * i2) * (logf(10000.0f) / (float)DMODEL));
  float ang = (float)l * div;
  acc += (dm & 1) ? cosf(ang) : sinf(ang);
  out[idx] = acc;
}

// ---------------- LayerNorm over last dim (512), one block per row ----
__global__ void ln_k(const float* __restrict__ x, const float* __restrict__ g,
                     const float* __restrict__ b, float* __restrict__ out)
{
  int row = blockIdx.x, tid = threadIdx.x;
  __shared__ float s1[256], s2[256];
  const float* xr = x + (long)row * DMODEL;
  float a0 = xr[tid], a1 = xr[tid + 256];
  s1[tid] = a0 + a1; s2[tid] = a0 * a0 + a1 * a1;
  __syncthreads();
  for (int st = 128; st > 0; st >>= 1) {
    if (tid < st) { s1[tid] += s1[tid + st]; s2[tid] += s2[tid + st]; }
    __syncthreads();
  }
  float mean = s1[0] * (1.0f / DMODEL);
  float var  = s2[0] * (1.0f / DMODEL) - mean * mean;
  float inv  = rsqrtf(var + 1e-5f);
  float* orow = out + (long)row * DMODEL;
  orow[tid]       = (a0 - mean) * inv * g[tid]       + b[tid];
  orow[tid + 256] = (a1 - mean) * inv * g[tid + 256] + b[tid + 256];
}

// ---------------- ProbSparse M measure -------------------------------
__global__ void prob_m_k(const float* __restrict__ q, const float* __restrict__ k,
                         float* __restrict__ Mout, int Lq, int Lk, int U, unsigned seed)
{
  long idx = (long)blockIdx.x * blockDim.x + threadIdx.x;
  if (idx >= (long)BHN * Lq) return;
  int lq = (int)(idx % Lq);
  int bh = (int)(idx / Lq);
  int h  = bh & (NHEADS - 1), b = bh >> 3;
  const float* qv = q + ((long)(b * Lq + lq) * DMODEL + h * DH);
  float mx = -1e30f, sm = 0.0f;
  for (int j = 0; j < U; ++j) {
    int r = (int)(rnd3(seed, (unsigned)lq, (unsigned)j) % (unsigned)Lk);
    const float* kv = k + ((long)(b * Lk + r) * DMODEL + h * DH);
    float d = 0.0f;
    for (int t = 0; t < DH; ++t) d += qv[t] * kv[t];
    mx = fmaxf(mx, d); sm += d;
  }
  Mout[idx] = mx - sm / (float)Lk;
}

// ---------------- iterative top-u (destroys M buffer) ----------------
__global__ void topk_k(float* __restrict__ Mv, int* __restrict__ top, int Lq, int u)
{
  int bb = blockIdx.x, tid = threadIdx.x;
  float* row = Mv + (long)bb * Lq;
  __shared__ float sv[256]; __shared__ int si[256];
  for (int it = 0; it < u; ++it) {
    float best = -3.0e38f; int bi = 0x7fffffff;
    for (int i = tid; i < Lq; i += 256) {
      float v = row[i];
      if (v > best || (v == best && i < bi)) { best = v; bi = i; }
    }
    sv[tid] = best; si[tid] = bi;
    __syncthreads();
    for (int st = 128; st > 0; st >>= 1) {
      if (tid < st) {
        if (sv[tid + st] > sv[tid] || (sv[tid + st] == sv[tid] && si[tid + st] < si[tid])) {
          sv[tid] = sv[tid + st]; si[tid] = si[tid + st];
        }
      }
      __syncthreads();
    }
    if (tid == 0) { top[bb * UPAD + it] = si[0]; row[si[0]] = -3.0e38f; }
    __syncthreads();
  }
}

// ---------------- gather selected q rows (zero-padded to UPAD) -------
__global__ void gather_q_k(const float* __restrict__ q, const int* __restrict__ top,
                           float* __restrict__ qp, int Lq, int u)
{
  long idx = (long)blockIdx.x * blockDim.x + threadIdx.x;
  if (idx >= (long)BHN * UPAD * DH) return;
  int d  = (int)(idx % DH);
  int r  = (int)((idx / DH) % UPAD);
  int bb = (int)(idx / ((long)DH * UPAD));
  float v = 0.0f;
  if (r < u) {
    int l = top[bb * UPAD + r];
    int b = bb >> 3, h = bb & (NHEADS - 1);
    v = q[(long)(b * Lq + l) * DMODEL + h * DH + d];
  }
  qp[idx] = v;
}

__global__ void causal_mask_k(float* __restrict__ sc, const int* __restrict__ top,
                              int u, int Lk, long total)
{
  long idx = (long)blockIdx.x * blockDim.x + threadIdx.x;
  if (idx >= total) return;
  int  lk = (int)(idx % Lk);
  long rr = idx / Lk;
  int  r  = (int)(rr % u);
  int  bb = (int)(rr / u);
  if (lk > top[bb * UPAD + r]) sc[((long)bb * UPAD + r) * Lk + lk] = NEGINF;
}

__global__ void softmax_k(float* __restrict__ sc, int u, int Lk)
{
  int row = blockIdx.x, bb = row / u, r = row % u, tid = threadIdx.x;
  float* p = sc + ((long)bb * UPAD + r) * Lk;
  __shared__ float s[256];
  float mx = -1e30f;
  for (int i = tid; i < Lk; i += 256) mx = fmaxf(mx, p[i]);
  s[tid] = mx; __syncthreads();
  for (int st = 128; st > 0; st >>= 1) { if (tid < st) s[tid] = fmaxf(s[tid], s[tid + st]); __syncthreads(); }
  mx = s[0]; __syncthreads();
  float sum = 0.0f;
  for (int i = tid; i < Lk; i += 256) { float e = expf(p[i] - mx); p[i] = e; sum += e; }
  s[tid] = sum; __syncthreads();
  for (int st = 128; st > 0; st >>= 1) { if (tid < st) s[tid] += s[tid + st]; __syncthreads(); }
  float inv = 1.0f / s[0];
  for (int i = tid; i < Lk; i += 256) p[i] *= inv;
}

__global__ void vmean_k(const float* __restrict__ v, float* __restrict__ vm, int Lk)
{
  int idx = blockIdx.x * blockDim.x + threadIdx.x;
  if (idx >= BSZ * DMODEL) return;
  int c = idx % DMODEL, b = idx / DMODEL;
  float s = 0.0f;
  for (int l = 0; l < Lk; ++l) s += v[(long)(b * Lk + l) * DMODEL + c];
  vm[idx] = s / (float)Lk;
}

__global__ void ctx_fill_k(const float* __restrict__ vm, float* __restrict__ ctx, int Lq)
{
  long idx = (long)blockIdx.x * blockDim.x + threadIdx.x;
  if (idx >= (long)BSZ * Lq * DMODEL) return;
  int  c  = (int)(idx % DMODEL);
  int  b  = (int)(idx / ((long)Lq * DMODEL));
  ctx[idx] = vm[b * DMODEL + c];
}

__global__ void ctx_cumsum_k(const float* __restrict__ v, float* __restrict__ ctx, int Lq)
{
  int idx = blockIdx.x * blockDim.x + threadIdx.x;
  if (idx >= BSZ * DMODEL) return;
  int c = idx % DMODEL, b = idx / DMODEL;
  float run = 0.0f;
  for (int l = 0; l < Lq; ++l) {
    long o = (long)(b * Lq + l) * DMODEL + c;
    run += v[o];
    ctx[o] = run;
  }
}

__global__ void scatter_k(const float* __restrict__ upd, const int* __restrict__ top,
                          float* __restrict__ ctx, int Lq, int u, long total)
{
  long idx = (long)blockIdx.x * blockDim.x + threadIdx.x;
  if (idx >= total) return;
  int  d  = (int)(idx % DH);
  long rr = idx / DH;
  int  r  = (int)(rr % u);
  int  bb = (int)(rr / u);
  int  b  = bb >> 3, h = bb & (NHEADS - 1);
  int  l  = top[bb * UPAD + r];
  ctx[(long)(b * Lq + l) * DMODEL + h * DH + d] = upd[((long)bb * UPAD + r) * DH + d];
}

// ---------------- distilling-conv BN stats / BN+ELU / maxpool --------
__global__ void bnstats_k(const float* __restrict__ y, float* __restrict__ st, int rows)
{
  int c = blockIdx.x, tid = threadIdx.x;
  __shared__ float s1[256], s2[256];
  float a = 0.0f, q = 0.0f;
  for (int r = tid; r < rows; r += 256) {
    float v = y[(long)r * DMODEL + c];
    a += v; q += v * v;
  }
  s1[tid] = a; s2[tid] = q; __syncthreads();
  for (int stp = 128; stp > 0; stp >>= 1) {
    if (tid < stp) { s1[tid] += s1[tid + stp]; s2[tid] += s2[tid + stp]; }
    __syncthreads();
  }
  if (tid == 0) {
    float m = s1[0] / (float)rows;
    st[c] = m;
    st[DMODEL + c] = s2[0] / (float)rows - m * m;
  }
}

__global__ void bnelu_k(float* __restrict__ y, const float* __restrict__ st,
                        const float* __restrict__ g, const float* __restrict__ be, long total)
{
  long idx = (long)blockIdx.x * blockDim.x + threadIdx.x;
  if (idx >= total) return;
  int c = (int)(idx % DMODEL);
  float v = (y[idx] - st[c]) * rsqrtf(st[DMODEL + c] + 1e-5f) * g[c] + be[c];
  y[idx] = v > 0.0f ? v : expm1f(v);
}

__global__ void maxpool_k(const float* __restrict__ y, float* __restrict__ out, int L)
{
  int Lo = L / 2;
  long idx = (long)blockIdx.x * blockDim.x + threadIdx.x;
  if (idx >= (long)BSZ * Lo * DMODEL) return;
  int  c  = (int)(idx % DMODEL);
  long bl = idx / DMODEL;
  int  lo = (int)(bl % Lo);
  int  b  = (int)(bl / Lo);
  float m = -1e30f;
#pragma unroll
  for (int t = 0; t < 3; ++t) {
    int li = 2 * lo - 1 + t;
    if (li >= 0 && li < L) m = fmaxf(m, y[(long)(b * L + li) * DMODEL + c]);
  }
  out[idx] = m;
}

__global__ void tail_k(const float* __restrict__ p, float* __restrict__ out)
{
  long idx = (long)blockIdx.x * blockDim.x + threadIdx.x;
  if (idx >= (long)BSZ * PREDL * C_OUT) return;
  int  c  = (int)(idx % C_OUT);
  long tl = idx / C_OUT;
  int  t  = (int)(tl % PREDL);
  int  b  = (int)(tl / PREDL);
  out[idx] = p[(long)(b * L_DEC + (L_DEC - PREDL + t)) * C_OUT + c];
}

// =====================================================================
// Host orchestration
// =====================================================================
struct AttnP { const float *Wq,*bq,*Wk,*bk,*Wv,*bv,*Wo,*bo; };

static inline int iceilln(int L) { return (int)std::ceil(std::log((double)L)); }

extern "C" void kernel_launch(void* const* d_in, const int* in_sizes, int n_in,
                              void* d_out, int out_size, void* d_ws, size_t ws_size,
                              hipStream_t stream) {
  (void)in_sizes; (void)n_in; (void)out_size; (void)ws_size;
  const float* x_enc    = (const float*)d_in[0];
  const float* mark_enc = (const float*)d_in[1];
  const float* x_dec    = (const float*)d_in[2];
  const float* mark_dec = (const float*)d_in[3];

  int pi = 4;
  auto P = [&]() { return (const float*)d_in[pi++]; };
  auto getAttn = [&]() {
    AttnP a;
    a.Wq = P(); a.bq = P(); a.Wk = P(); a.bk = P();
    a.Wv = P(); a.bv = P(); a.Wo = P(); a.bo = P();
    return a;
  };

  struct Emb { const float *convW,*timeW,*timeb; };
  Emb eemb; eemb.convW = P(); eemb.timeW = P(); eemb.timeb = P();
  Emb demb; demb.convW = P(); demb.timeW = P(); demb.timeb = P();

  struct EncL { AttnP a; const float *g1,*b1,*W1,*c1,*W2,*c2,*g2,*b2; };
  EncL encl[3];
  for (int i = 0; i < 3; ++i) {
    encl[i].a = getAttn();
    encl[i].g1 = P(); encl[i].b1 = P();
    encl[i].W1 = P(); encl[i].c1 = P();
    encl[i].W2 = P(); encl[i].c2 = P();
    encl[i].g2 = P(); encl[i].b2 = P();
  }
  struct ConvP { const float *W,*b,*g,*beta; };
  ConvP cv[2];
  for (int i = 0; i < 2; ++i) { cv[i].W = P(); cv[i].b = P(); cv[i].g = P(); cv[i].beta = P(); }
  const float* enorm_g = P(); const float* enorm_b = P();
  struct DecL { AttnP sa, ca; const float *g1,*b1,*g2,*b2,*W1,*c1,*W2,*c2,*g3,*b3; };
  DecL decl[2];
  for (int i = 0; i < 2; ++i) {
    decl[i].sa = getAttn();
    decl[i].ca = getAttn();
    decl[i].g1 = P(); decl[i].b1 = P(); decl[i].g2 = P(); decl[i].b2 = P();
    decl[i].W1 = P(); decl[i].c1 = P(); decl[i].W2 = P(); decl[i].c2 = P();
    decl[i].g3 = P(); decl[i].b3 = P();
  }
  const float* dnorm_g = P(); const float* dnorm_b = P();
  const float* projW = P();  const float* projb = P();

  // ---------------- workspace carve-out ----------------
  char* wsb = (char*)d_ws;
  size_t cur = 0;
  auto alloc = [&](size_t bytes) {
    void* p = wsb + cur;
    cur += (bytes + 255) & ~(size_t)255;
    return p;
  };
  const long ROWS = (long)BSZ * L_ENC;
  float* X   = (float*)alloc(ROWS * DMODEL * 4);
  float* Yb  = (float*)alloc(ROWS * DMODEL * 4);
  float* Qb  = (float*)alloc(ROWS * DMODEL * 4);
  float* Kb  = (float*)alloc(ROWS * DMODEL * 4);
  float* Vb  = (float*)alloc(ROWS * DMODEL * 4);
  float* Hb  = (float*)alloc(ROWS * DFF * 4);
  float* CTX = (float*)alloc(ROWS * DMODEL * 4);
  float* SC  = (float*)alloc((long)BHN * UPAD * L_ENC * 4);
  float* QP  = (float*)alloc((long)BHN * UPAD * DH * 4);
  float* UD  = (float*)alloc((long)BHN * UPAD * DH * 4);
  float* MB  = (float*)alloc((long)BHN * L_ENC * 4);
  int*   TOP = (int*)  alloc((long)BHN * UPAD * 4);
  float* VM  = (float*)alloc((long)BSZ * DMODEL * 4);
  float* ST  = (float*)alloc(2 * DMODEL * 4);
  float* ENC = (float*)alloc((long)BSZ * 512 * DMODEL * 4);
  float* PR  = (float*)alloc((long)BSZ * L_DEC * 8 * 4);

  dim3 blk(256);

  // dense GEMM dispatch: tiled async-LDS kernel when shape allows
  auto gemm = [&](const float* A, const float* W, const float* bias, const float* res,
                  float* C, int M, int N, int K, int act) {
    if ((M % BM == 0) && (N % BN == 0) && (K % BK == 0)) {
      dim3 g((M / BM) * (N / BN), 1);
      if (act)
        wmma_gemm_tiled_k<1, false><<<g, blk, 0, stream>>>(A, W, bias, nullptr, C, M, N, K);
      else if (res)
        wmma_gemm_tiled_k<0, true><<<g, blk, 0, stream>>>(A, W, bias, res, C, M, N, K);
      else
        wmma_gemm_tiled_k<0, false><<<g, blk, 0, stream>>>(A, W, bias, nullptr, C, M, N, K);
      return;
    }
    int tiles = ((M + 15) / 16) * ((N + 15) / 16);
    dim3 g((tiles + 7) / 8, 1);
    if (act)
      wmma_gemm_k<1, true, false, false><<<g, blk, 0, stream>>>(
        A, W, bias, nullptr, C, M, N, K, K, N, N, 0, 0, 0, 0, 0, 0, 1, tiles, 1.0f);
    else if (res)
      wmma_gemm_k<0, true, true, false><<<g, blk, 0, stream>>>(
        A, W, bias, res, C, M, N, K, K, N, N, 0, 0, 0, 0, 0, 0, 1, tiles, 1.0f);
    else
      wmma_gemm_k<0, true, false, false><<<g, blk, 0, stream>>>(
        A, W, bias, nullptr, C, M, N, K, K, N, N, 0, 0, 0, 0, 0, 0, 1, tiles, 1.0f);
  };

  auto lnorm = [&](const float* x, const float* g, const float* b, float* out, int rows) {
    ln_k<<<rows, blk, 0, stream>>>(x, g, b, out);
  };

  // ProbSparse attention; writes (xq + attn_out) into outb
  auto attention = [&](const float* xq, const float* xkv, int Lq, int Lk,
                       const AttnP& p, bool mask, unsigned seed, float* outb) {
    int Mq = BSZ * Lq, Mk = BSZ * Lk;
    gemm(xq,  p.Wq, p.bq, nullptr, Qb, Mq, DMODEL, DMODEL, 0);
    gemm(xkv, p.Wk, p.bk, nullptr, Kb, Mk, DMODEL, DMODEL, 0);
    gemm(xkv, p.Wv, p.bv, nullptr, Vb, Mk, DMODEL, DMODEL, 0);

    int U = std::min(5 * iceilln(Lk), Lk);
    int u = std::min(5 * iceilln(Lq), Lq);

    long t1 = (long)BHN * Lq;
    prob_m_k<<<(int)((t1 + 255) / 256), blk, 0, stream>>>(Qb, Kb, MB, Lq, Lk, U, seed);
    topk_k<<<BHN, blk, 0, stream>>>(MB, TOP, Lq, u);
    long t2 = (long)BHN * UPAD * DH;
    gather_q_k<<<(int)((t2 + 255) / 256), blk, 0, stream>>>(Qb, TOP, QP, Lq, u);

    // scores = (1/sqrt(DH)) * q_r @ k^T  : batched over (b,h)
    {
      int tiles = (UPAD / 16) * ((Lk + 15) / 16);
      dim3 g((tiles + 7) / 8, BHN);
      wmma_gemm_k<0, false, false, true><<<g, blk, 0, stream>>>(
        QP, Kb, nullptr, nullptr, SC,
        UPAD, Lk, DH, DH, DMODEL, Lk,
        (long)NHEADS * UPAD * DH, (long)UPAD * DH,
        (long)Lk * DMODEL, (long)DH,
        (long)NHEADS * UPAD * Lk, (long)UPAD * Lk,
        NHEADS, tiles, 0.125f);
    }
    if (mask) {
      long t3 = (long)BHN * u * Lk;
      causal_mask_k<<<(int)((t3 + 255) / 256), blk, 0, stream>>>(SC, TOP, u, Lk, t3);
    }
    softmax_k<<<BHN * u, blk, 0, stream>>>(SC, u, Lk);

    // context init
    if (mask) {
      ctx_cumsum_k<<<(BSZ * DMODEL + 255) / 256, blk, 0, stream>>>(Vb, CTX, Lq);
    } else {
      vmean_k<<<(BSZ * DMODEL + 255) / 256, blk, 0, stream>>>(Vb, VM, Lk);
      long t4 = (long)BSZ * Lq * DMODEL;
      ctx_fill_k<<<(int)((t4 + 255) / 256), blk, 0, stream>>>(VM, CTX, Lq);
    }

    // upd = attn @ v : batched
    {
      int tiles = (UPAD / 16) * (DH / 16);
      dim3 g((tiles + 7) / 8, BHN);
      wmma_gemm_k<0, false, false, false><<<g, blk, 0, stream>>>(
        SC, Vb, nullptr, nullptr, UD,
        UPAD, DH, Lk, Lk, DMODEL, DH,
        (long)NHEADS * UPAD * Lk, (long)UPAD * Lk,
        (long)Lk * DMODEL, (long)DH,
        (long)NHEADS * UPAD * DH, (long)UPAD * DH,
        NHEADS, tiles, 1.0f);
    }
    long t5 = (long)BHN * u * DH;
    scatter_k<<<(int)((t5 + 255) / 256), blk, 0, stream>>>(UD, TOP, CTX, Lq, u, t5);

    // out = ctx @ Wo + bo + xq
    gemm(CTX, p.Wo, p.bo, xq, outb, Mq, DMODEL, DMODEL, 0);
  };

  // ======================= Encoder =======================
  int L = L_ENC;
  {
    long te = (long)BSZ * L * DMODEL;
    embed_k<<<(int)((te + 255) / 256), blk, 0, stream>>>(
      x_enc, mark_enc, eemb.convW, eemb.timeW, eemb.timeb, X, L);
  }
  unsigned rng = 0;
  for (int i = 0; i < 3; ++i) {
    int rows = BSZ * L;
    attention(X, X, L, L, encl[i].a, false, rng, Yb); rng++;
    lnorm(Yb, encl[i].g1, encl[i].b1, X, rows);
    gemm(X, encl[i].W1, encl[i].c1, nullptr, Hb, rows, DFF, DMODEL, 1);   // GELU
    gemm(Hb, encl[i].W2, encl[i].c2, X, Yb, rows, DMODEL, DFF, 0);        // +resid
    lnorm(Yb, encl[i].g2, encl[i].b2, X, rows);
    if (i < 2) {
      int tiles = (rows / 16) * (DMODEL / 16);
      wmma_convgemm_k<<<(tiles + 7) / 8, blk, 0, stream>>>(X, cv[i].W, cv[i].b, Yb, L, tiles);
      bnstats_k<<<DMODEL, blk, 0, stream>>>(Yb, ST, rows);
      long tb = (long)rows * DMODEL;
      bnelu_k<<<(int)((tb + 255) / 256), blk, 0, stream>>>(Yb, ST, cv[i].g, cv[i].beta, tb);
      long tp = (long)BSZ * (L / 2) * DMODEL;
      maxpool_k<<<(int)((tp + 255) / 256), blk, 0, stream>>>(Yb, X, L);
      L /= 2;
    }
  }
  lnorm(X, enorm_g, enorm_b, ENC, BSZ * L);   // enc_out, L == 512
  const int Le = L;

  // ======================= Decoder =======================
  {
    long td = (long)BSZ * L_DEC * DMODEL;
    embed_k<<<(int)((td + 255) / 256), blk, 0, stream>>>(
      x_dec, mark_dec, demb.convW, demb.timeW, demb.timeb, X, L_DEC);
  }
  for (int i = 0; i < 2; ++i) {
    int rows = BSZ * L_DEC;
    attention(X, X, L_DEC, L_DEC, decl[i].sa, true, rng, Yb); rng++;
    lnorm(Yb, decl[i].g1, decl[i].b1, X, rows);
    attention(X, ENC, L_DEC, Le, decl[i].ca, false, rng, Yb); rng++;
    lnorm(Yb, decl[i].g2, decl[i].b2, X, rows);
    gemm(X, decl[i].W1, decl[i].c1, nullptr, Hb, rows, DFF, DMODEL, 1);
    gemm(Hb, decl[i].W2, decl[i].c2, X, Yb, rows, DMODEL, DFF, 0);
    lnorm(Yb, decl[i].g3, decl[i].b3, X, rows);
  }
  lnorm(X, dnorm_g, dnorm_b, X, BSZ * L_DEC);
  gemm(X, projW, projb, nullptr, PR, BSZ * L_DEC, C_OUT, DMODEL, 0);
  {
    long tt = (long)BSZ * PREDL * C_OUT;
    tail_k<<<(int)((tt + 255) / 256), blk, 0, stream>>>(PR, (float*)d_out);
  }
}
